// MultiHeadSelfAttention_8057358647638
// MI455X (gfx1250) — matrix operand order repaired and checked by reading, not compiled, not run on hardware
//
#include <hip/hip_runtime.h>

#define B_  4
#define C_  256
#define HW_ 4096
#define KPAD 264            // 256 + 8 bf16 pad (16B) -> conflict-free fragment reads
#define VPAD 40             // 32 + 8
#define PPAD 40             // 32 + 8

typedef __bf16 bf16x8  __attribute__((ext_vector_type(8)));
typedef __bf16 bf16x16 __attribute__((ext_vector_type(16)));
typedef float  floatx8 __attribute__((ext_vector_type(8)));

union AF { bf16x16 v; bf16x8 h[2]; };

__device__ inline floatx8 zero8() {
  floatx8 z = {0.f,0.f,0.f,0.f,0.f,0.f,0.f,0.f};
  return z;
}

__device__ inline __bf16 f2bf(float f) {
  unsigned u = __builtin_bit_cast(unsigned, f);
  u = (u + 0x7FFFu + ((u >> 16) & 1u)) >> 16;     // round-to-nearest-even
  unsigned short s = (unsigned short)u;
  return __builtin_bit_cast(__bf16, s);
}

__device__ inline floatx8 wmma_bf16(bf16x16 a, bf16x16 b, floatx8 c) {
  return __builtin_amdgcn_wmma_f32_16x16x32_bf16(false, a, false, b, (short)0, c, false, false);
}

// Async global->LDS copy, 16B per lane.  GV mode: vdst = per-lane LDS byte
// offset, vaddr = 64-bit global address.  Tracked by ASYNCcnt.
__device__ inline void async_b128(unsigned lds_byte_off, const void* gaddr) {
  asm volatile("global_load_async_to_lds_b128 %0, %1, off"
               :: "v"(lds_byte_off), "v"((unsigned long long)(size_t)gaddr)
               : "memory");
}
#define WAIT_ASYNC(n) asm volatile("s_wait_asynccnt %0" :: "i"(n) : "memory")

__device__ inline unsigned lds_off(const void* p) {
  return (unsigned)(size_t)p;   // generic LDS pointer: low 32 bits = LDS offset
}

// ---------------------------------------------------------------------------
// Kernel 1: xt[b][q][c] = x[b][c][q] + pe(c,h,w), bf16, [b,hw,c]
// ---------------------------------------------------------------------------
__global__ void k_pe_add(const float* __restrict__ x, __bf16* __restrict__ xt) {
  int idx = blockIdx.x * blockDim.x + threadIdx.x;
  int c = idx & (C_ - 1);
  int q = (idx >> 8) & (HW_ - 1);
  int b = idx >> 20;
  int hpos = q >> 6, wpos = q & 63;
  int i = (c & 127) >> 1;
  float divv = __expf(-(float)(2 * i) * (9.2103403719761836f / 128.0f));
  float pos  = (c < 128) ? (float)wpos : (float)hpos;
  float arg  = pos * divv;
  float pe   = (c & 1) ? __cosf(arg) : __sinf(arg);
  float v    = x[((size_t)(b * C_ + c) * HW_) + q] + pe;
  xt[(size_t)(b * HW_ + q) * C_ + c] = f2bf(v);
}

// ---------------------------------------------------------------------------
// Kernel 2: Wt[w][n][k] = W[k][n]  (bf16, transposed for B-fragment loads)
// ---------------------------------------------------------------------------
__global__ void k_wt(const float* __restrict__ Wq, const float* __restrict__ Wk,
                     const float* __restrict__ Wv, __bf16* __restrict__ wt) {
  int idx = blockIdx.x * blockDim.x + threadIdx.x;
  int k = idx & (C_ - 1);
  int n = (idx >> 8) & (C_ - 1);
  int w = idx >> 16;
  const float* src = (w == 0) ? Wq : (w == 1) ? Wk : Wv;
  wt[(size_t)w * C_ * C_ + n * C_ + k] = f2bf(src[k * C_ + n]);
}

// ---------------------------------------------------------------------------
// Kernel 3: QKV projections; wave -> 16x64 tile.  Q,K row-major; V transposed.
// ---------------------------------------------------------------------------
__global__ void __launch_bounds__(256)
k_qkv(const __bf16* __restrict__ xt, const __bf16* __restrict__ wt,
      __bf16* __restrict__ Qm, __bf16* __restrict__ Km, __bf16* __restrict__ Vt) {
  int lane = threadIdx.x & 31, wave = threadIdx.x >> 5;
  int laneLo = lane & 15, laneHi = lane >> 4;
  int b = blockIdx.z / 3, widx = blockIdx.z % 3;
  const __bf16* A  = xt + (size_t)b * HW_ * C_;
  const __bf16* Bw = wt + (size_t)widx * C_ * C_;
  int m0 = blockIdx.x * 128 + wave * 16;
  int n0 = blockIdx.y * 64;
  floatx8 acc[4];
  for (int t = 0; t < 4; ++t) acc[t] = zero8();
  int rowA = m0 + laneLo;
  for (int k0 = 0; k0 < C_; k0 += 32) {
    AF af;
    const __bf16* ap = A + (size_t)rowA * C_ + k0 + laneHi * 8;
    af.h[0] = *(const bf16x8*)ap;
    af.h[1] = *(const bf16x8*)(ap + 16);
    for (int t = 0; t < 4; ++t) {
      AF bf;
      const __bf16* bp = Bw + (size_t)(n0 + t * 16 + laneLo) * C_ + k0 + laneHi * 8;
      bf.h[0] = *(const bf16x8*)bp;
      bf.h[1] = *(const bf16x8*)(bp + 16);
      acc[t] = wmma_bf16(af.v, bf.v, acc[t]);
    }
  }
  if (widx < 2) {
    __bf16* Out = (widx == 0 ? Qm : Km) + (size_t)b * HW_ * C_;
    for (int t = 0; t < 4; ++t)
      for (int r = 0; r < 8; ++r)
        Out[(size_t)(m0 + r + laneHi * 8) * C_ + n0 + t * 16 + laneLo] = f2bf(acc[t][r]);
  } else {
    __bf16* Out = Vt + (size_t)b * HW_ * C_;
    for (int t = 0; t < 4; ++t)
      for (int r = 0; r < 8; ++r)
        Out[(size_t)(n0 + t * 16 + laneLo) * HW_ + m0 + r + laneHi * 8] = f2bf(acc[t][r]);
  }
}

// ---------------------------------------------------------------------------
// Kernel 4: per-key softmax stats.  Wave owns 16 keys (K B-fragments hoisted
// to registers); Q tiles streamed through double-buffered LDS via async copy.
// ---------------------------------------------------------------------------
__global__ void __launch_bounds__(256)
k_stats(const __bf16* __restrict__ Qm, const __bf16* __restrict__ Km,
        float* __restrict__ colmax, float* __restrict__ colsum) {
  __shared__ __attribute__((aligned(16))) __bf16 Qb2[2][16 * KPAD];
  int tid = threadIdx.x;
  int lane = tid & 31, wave = tid >> 5;
  int laneLo = lane & 15, laneHi = lane >> 4;
  int b  = blockIdx.y;
  int kb = blockIdx.x * 128 + wave * 16;
  const __bf16* Q = Qm + (size_t)b * HW_ * C_;
  const __bf16* K = Km + (size_t)b * HW_ * C_;
  const float scale = 0.0625f;

  // hoist K B-fragments (loop-invariant): 8 k-steps x 8 VGPRs
  AF kf[8];
  for (int k = 0; k < 8; ++k) {
    const __bf16* bp = K + (size_t)(kb + laneLo) * C_ + k * 32 + laneHi * 8;
    kf[k].h[0] = *(const bf16x8*)bp;
    kf[k].h[1] = *(const bf16x8*)(bp + 16);
  }

  // staging split: thread t copies row t/16, 16-elem chunk (t%16) -> 32B
  int qrow = tid >> 4, qchunk = (tid & 15) * 16;
  auto stage = [&](int buf, int q0) {
    const __bf16* src = Q + (size_t)(q0 + qrow) * C_ + qchunk;
    unsigned dst = lds_off(&Qb2[buf][qrow * KPAD + qchunk]);
    async_b128(dst,      src);
    async_b128(dst + 16, src + 8);
  };

  stage(0, 0);
  float m_run = -3.0e38f, z_run = 0.0f;
  int cur = 0;
  for (int q0 = 0; q0 < HW_; q0 += 16, cur ^= 1) {
    if (q0 + 16 < HW_) { stage(cur ^ 1, q0 + 16); WAIT_ASYNC(2); }
    else               { WAIT_ASYNC(0); }
    __syncthreads();
    const __bf16* Qc = &Qb2[cur][0];
    floatx8 acc = zero8();
    for (int k = 0; k < 8; ++k) {
      AF af;
      const __bf16* ap = Qc + laneLo * KPAD + k * 32 + laneHi * 8;
      af.h[0] = *(const bf16x8*)ap;
      af.h[1] = *(const bf16x8*)(ap + 16);
      acc = wmma_bf16(af.v, kf[k].v, acc);
    }
    float tmax = acc[0] * scale;
    for (int r = 1; r < 8; ++r) tmax = fmaxf(tmax, acc[r] * scale);
    tmax = fmaxf(tmax, __shfl_xor(tmax, 16, 32));
    float new_m = fmaxf(m_run, tmax);
    float p = 0.0f;
    for (int r = 0; r < 8; ++r) p += __expf(acc[r] * scale - new_m);
    p += __shfl_xor(p, 16, 32);
    z_run = z_run * __expf(m_run - new_m) + p;
    m_run = new_m;
    __syncthreads();
  }
  if (laneHi == 0) {
    colmax[b * HW_ + kb + laneLo] = m_run;
    colsum[b * HW_ + kb + laneLo] = z_run;
  }
}

// ---------------------------------------------------------------------------
// Kernel 5: output pass.  Q A-fragments hoisted to registers; K/V key-blocks
// streamed through double-buffered LDS (async copy, shared by all 8 waves);
// P transposed through padded per-wave LDS; P @ V via WMMA.
// ---------------------------------------------------------------------------
__global__ void __launch_bounds__(256)
k_out(const __bf16* __restrict__ Qm, const __bf16* __restrict__ Km,
      const __bf16* __restrict__ Vt, const float* __restrict__ colmax,
      const float* __restrict__ colsum, float* __restrict__ out) {
  __shared__ __attribute__((aligned(16))) __bf16 Kb2[2][32 * KPAD];
  __shared__ __attribute__((aligned(16))) __bf16 Vb2[2][256 * VPAD];
  __shared__ __attribute__((aligned(16))) __bf16 Pbuf[8][16 * PPAD];
  int tid = threadIdx.x;
  int lane = tid & 31, wave = tid >> 5;
  int laneLo = lane & 15, laneHi = lane >> 4;
  int b  = blockIdx.y;
  int q0 = blockIdx.x * 128 + wave * 16;
  const __bf16* Q = Qm + (size_t)b * HW_ * C_;
  const __bf16* K = Km + (size_t)b * HW_ * C_;
  const __bf16* V = Vt + (size_t)b * HW_ * C_;
  const float* cm = colmax + b * HW_;
  const float* cs = colsum + b * HW_;
  const float scale = 0.0625f;

  // hoist Q A-fragments (loop-invariant across keys): 8 x 8 VGPRs
  AF qf[8];
  {
    int rowA = q0 + laneLo;
    for (int k = 0; k < 8; ++k) {
      const __bf16* ap = Q + (size_t)rowA * C_ + k * 32 + laneHi * 8;
      qf[k].h[0] = *(const bf16x8*)ap;
      qf[k].h[1] = *(const bf16x8*)(ap + 16);
    }
  }

  // staging split: K block 32x256 (row t/8, 64B chunk t%8); V block 256x32
  int krow = tid >> 3, kchunk = (tid & 7) * 32;
  int vrow = tid;
  auto stage = [&](int buf, int kb) {
    const __bf16* kg = K + (size_t)(kb + krow) * C_ + kchunk;
    unsigned kl = lds_off(&Kb2[buf][krow * KPAD + kchunk]);
    const __bf16* vg = V + (size_t)vrow * HW_ + kb;
    unsigned vl = lds_off(&Vb2[buf][vrow * VPAD]);
    for (int j = 0; j < 4; ++j) async_b128(kl + j * 16, kg + j * 8);
    for (int j = 0; j < 4; ++j) async_b128(vl + j * 16, vg + j * 8);
  };

  stage(0, 0);
  floatx8 acc[16];
  for (int t = 0; t < 16; ++t) acc[t] = zero8();
  int cur = 0;
  for (int kb = 0; kb < HW_; kb += 32, cur ^= 1) {
    if (kb + 32 < HW_) { stage(cur ^ 1, kb + 32); WAIT_ASYNC(8); }
    else               { WAIT_ASYNC(0); }
    __syncthreads();
    const __bf16* Kc = &Kb2[cur][0];
    const __bf16* Vc = &Vb2[cur][0];

    // S tiles: 16 queries x 32 keys
    floatx8 s0 = zero8(), s1 = zero8();
    for (int k = 0; k < 8; ++k) {
      AF b0, b1;
      const __bf16* bp0 = Kc + laneLo * KPAD + k * 32 + laneHi * 8;
      b0.h[0] = *(const bf16x8*)bp0; b0.h[1] = *(const bf16x8*)(bp0 + 16);
      const __bf16* bp1 = Kc + (16 + laneLo) * KPAD + k * 32 + laneHi * 8;
      b1.h[0] = *(const bf16x8*)bp1; b1.h[1] = *(const bf16x8*)(bp1 + 16);
      s0 = wmma_bf16(qf[k].v, b0.v, s0);
      s1 = wmma_bf16(qf[k].v, b1.v, s1);
    }

    // P = exp(S - m_k) / Z_k, transpose D-layout -> A-layout through LDS
    float mA = cm[kb + laneLo],      rzA = 1.0f / cs[kb + laneLo];
    float mB = cm[kb + 16 + laneLo], rzB = 1.0f / cs[kb + 16 + laneLo];
    __bf16* pb = &Pbuf[wave][0];
    for (int r = 0; r < 8; ++r) {
      int m = r + laneHi * 8;
      pb[m * PPAD + laneLo]      = f2bf(__expf(s0[r] * scale - mA) * rzA);
      pb[m * PPAD + 16 + laneLo] = f2bf(__expf(s1[r] * scale - mB) * rzB);
    }
    asm volatile("" ::: "memory");   // keep ds_store before ds_load (same wave, HW in-order)
    AF pf;
    const __bf16* pp = pb + laneLo * PPAD + laneHi * 8;
    pf.h[0] = *(const bf16x8*)pp;
    pf.h[1] = *(const bf16x8*)(pp + 16);

    // out += P @ V
    for (int t = 0; t < 16; ++t) {
      AF vf;
      const __bf16* vp = Vc + (size_t)(t * 16 + laneLo) * VPAD + laneHi * 8;
      vf.h[0] = *(const bf16x8*)vp;
      vf.h[1] = *(const bf16x8*)(vp + 16);
      acc[t] = wmma_bf16(pf.v, vf.v, acc[t]);
    }
    __syncthreads();
  }

  float* ob = out + (size_t)b * C_ * HW_;
  for (int t = 0; t < 16; ++t)
    for (int r = 0; r < 8; ++r)
      ob[(size_t)(t * 16 + laneLo) * HW_ + q0 + r + laneHi * 8] = acc[t][r];
}

// ---------------------------------------------------------------------------
extern "C" void kernel_launch(void* const* d_in, const int* in_sizes, int n_in,
                              void* d_out, int out_size, void* d_ws, size_t ws_size,
                              hipStream_t stream) {
  const float* x  = (const float*)d_in[0];
  const float* Wq = (const float*)d_in[1];
  const float* Wk = (const float*)d_in[2];
  const float* Wv = (const float*)d_in[3];
  float* out = (float*)d_out;

  char* ws = (char*)d_ws;
  size_t off = 0;
  __bf16* XT = (__bf16*)(ws + off); off += (size_t)B_ * HW_ * C_ * 2;
  __bf16* Qm = (__bf16*)(ws + off); off += (size_t)B_ * HW_ * C_ * 2;
  __bf16* Km = (__bf16*)(ws + off); off += (size_t)B_ * HW_ * C_ * 2;
  __bf16* Vt = (__bf16*)(ws + off); off += (size_t)B_ * HW_ * C_ * 2;
  __bf16* Wt = (__bf16*)(ws + off); off += (size_t)3 * C_ * C_ * 2;
  float* colmax = (float*)(ws + off); off += (size_t)B_ * HW_ * 4;
  float* colsum = (float*)(ws + off); off += (size_t)B_ * HW_ * 4;
  (void)ws_size; (void)in_sizes; (void)n_in; (void)out_size;

  k_pe_add<<<(B_ * HW_ * C_) / 256, 256, 0, stream>>>(x, XT);
  k_wt<<<(3 * C_ * C_) / 256, 256, 0, stream>>>(Wq, Wk, Wv, Wt);
  k_qkv<<<dim3(HW_ / 128, C_ / 64, B_ * 3), 256, 0, stream>>>(XT, Wt, Qm, Km, Vt);
  k_stats<<<dim3(HW_ / 128, B_), 256, 0, stream>>>(Qm, Km, colmax, colsum);
  k_out<<<dim3(HW_ / 128, B_), 256, 0, stream>>>(Qm, Km, Vt, colmax, colsum, out);
}
